// DiffMamba_65730179498079
// MI455X (gfx1250) — matrix operand order repaired and checked by reading, not compiled, not run on hardware
//
#include <hip/hip_runtime.h>
#include <hip/hip_bf16.h>

#define D_MODEL   256
#define HEADDIM   16
#define NH        16
#define D_STATE   64
#define D_CONV    4
#define D_INNER   32
#define NH_IN     2
#define CONV_DIM  160          // D_INNER + 2*D_STATE
#define D_IN_PROJ 194          // 2*D_INNER + 2*D_STATE + NH_IN
#define BSZ       2
#define TLEN      2048
#define CHUNK     32
#define NCHUNK    (TLEN / CHUNK)
#define NPAD      208          // 13 * 16 tiles covering 194 cols

typedef _Float16 v16h __attribute__((ext_vector_type(16)));
typedef _Float16 v8h  __attribute__((ext_vector_type(8)));
typedef float    v8f  __attribute__((ext_vector_type(8)));

__device__ __forceinline__ v8f zero8() {
    v8f z;
#pragma unroll
    for (int i = 0; i < 8; ++i) z[i] = 0.0f;
    return z;
}
__device__ __forceinline__ v8h zero8h() {
    v8h z;
#pragma unroll
    for (int i = 0; i < 8; ++i) z[i] = (_Float16)0.0f;
    return z;
}
__device__ __forceinline__ v16h cat8(v8h lo, v8h hi) {
    return __builtin_shufflevector(lo, hi, 0,1,2,3,4,5,6,7,8,9,10,11,12,13,14,15);
}

// ---- WMMA operand loaders: all contiguous 16B vector loads ----------------
// A-matrix 16x32 f16, row-major (m,k) at p[m*ld+k].  Lane layout:
// element j -> K=kb+j, kb = (lane<16)?0:16.
__device__ __forceinline__ v16h loadA16(const _Float16* p, int ld, int lane) {
    int m  = lane & 15;
    int kb = (lane < 16) ? 0 : 16;
    const _Float16* q = p + m * ld + kb;
    v8h lo = *(const v8h*)(q);
    v8h hi = *(const v8h*)(q + 8);
    return cat8(lo, hi);
}
// A with only K<16 valid: upper half is a compile-time zero vector.
__device__ __forceinline__ v16h loadA16_k16(const _Float16* p, int ld, int lane) {
    v8h lo = zero8h(), hi = zero8h();
    if (lane < 16) {
        const _Float16* q = p + (lane & 15) * ld;
        lo = *(const v8h*)(q);
        hi = *(const v8h*)(q + 8);
    }
    return cat8(lo, hi);
}
// B-matrix 32x16 f16 in "B-transposed" layout: element (k,n) at p[n*ld+k].
// Lane layout: element j -> K=kb+j, kb = (lane<16)?0:16.
__device__ __forceinline__ v16h loadBT16(const _Float16* p, int ld, int lane) {
    int n  = lane & 15;
    int kb = (lane < 16) ? 0 : 16;
    const _Float16* q = p + n * ld + kb;
    v8h lo = *(const v8h*)(q);
    v8h hi = *(const v8h*)(q + 8);
    return cat8(lo, hi);
}
// BT with only K<16 valid (single exec-masked vector load, not 16 branches)
__device__ __forceinline__ v16h loadBT16_k16(const _Float16* p, int ld, int lane) {
    v8h lo = zero8h(), hi = zero8h();
    if (lane < 16) {
        const _Float16* q = p + (lane & 15) * ld;
        lo = *(const v8h*)(q);
        hi = *(const v8h*)(q + 8);
    }
    return cat8(lo, hi);
}
// BT from f32 source (contiguous, converts)
__device__ __forceinline__ v16h loadBT16f(const float* p, int ld, int lane) {
    int n  = lane & 15;
    int kb = (lane < 16) ? 0 : 16;
    const float* q = p + n * ld + kb;
    v16h b;
#pragma unroll
    for (int j = 0; j < 16; ++j) b[j] = (_Float16)q[j];
    return b;
}
// C/D f32 tile: element r -> row mb+r, col n
__device__ __forceinline__ v8f loadC8(const float* q, int ld, int lane) {
    int n = lane & 15, mb = (lane < 16) ? 0 : 8;
    v8f c;
#pragma unroll
    for (int r = 0; r < 8; ++r) c[r] = q[(mb + r) * ld + n];
    return c;
}

#define WMMA_F16(a, b, c) \
    __builtin_amdgcn_wmma_f32_16x16x32_f16(false, (a), false, (b), (short)0, (c), false, false)

__device__ __forceinline__ float sigmoidf_(float v) { return 1.0f / (1.0f + __expf(-v)); }
__device__ __forceinline__ float siluf_(float v)    { return v * sigmoidf_(v); }
__device__ __forceinline__ float softplusf_(float v){ return (v > 20.0f) ? v : log1pf(__expf(v)); }

// ===========================================================================
// Kernel 1: fused Mamba2 forward (chunked SSD, all GEMMs on WMMA).
// grid = NH*BSZ, block = 256 (8 waves).
// ===========================================================================
__global__ __launch_bounds__(256) void mamba_fwd_kernel(
    const float* __restrict__ x,      // [B,T,256]
    const float* __restrict__ Win,    // [NH,194,16]
    const float* __restrict__ convw,  // [NH,160,4]
    const float* __restrict__ convb,  // [NH,160]
    const float* __restrict__ dtb,    // [NH,2]
    const float* __restrict__ Alog,   // [NH,2]
    const float* __restrict__ Dp,     // [NH,2]
    const float* __restrict__ nw,     // [NH,32]
    const float* __restrict__ Wout,   // [NH,16,32]
    float* __restrict__ yout)         // [NH,B,T,16]
{
    __shared__ __align__(16) _Float16 sWinT[NPAD][16];     // Win rows (BT layout), rows>=194 zero
    __shared__ __align__(16) _Float16 sWoT[16][32];        // Wout rows (BT layout)
    __shared__ __align__(16) _Float16 sU16[CHUNK][16];     // u chunk (A, K=16)
    __shared__ __align__(16) _Float16 sZX16[CHUNK][NPAD];  // in_proj output
    __shared__ __align__(16) _Float16 sConv16[CHUNK][CONV_DIM]; // silu(conv(xBC))
    __shared__ __align__(16) _Float16 sXtail[3][CONV_DIM]; // raw xBC tail
    __shared__ __align__(16) _Float16 sXdtT[16][CHUNK];    // (dt*x)ᵀ [p][s]  (BT layout)
    __shared__ __align__(16) _Float16 sBd16[D_STATE][40];  // decayed Bᵀ [n][s] (A operand)
    __shared__ __align__(16) _Float16 sGm[CHUNK][40];      // masked C·Bᵀ (A operand)
    __shared__ __align__(16) float    sHstT[NH_IN][16][D_STATE]; // state hᵀ[p][n], f32
    __shared__ __align__(16) float    sYF[CHUNK][D_INNER]; // y accumulation, f32
    __shared__ __align__(16) _Float16 sYg16[CHUNK][D_INNER];// gated+normed (A)
    __shared__ __align__(16) float    sP[NH_IN][CHUNK];
    __shared__ __align__(16) float    sdt[NH_IN][CHUNK];
    __shared__ __align__(16) _Float16 sConvW[CONV_DIM][4];
    __shared__ __align__(16) _Float16 sConvB[CONV_DIM];
    __shared__ float sPl[NH_IN], sA[NH_IN], sDtb[NH_IN], sDpv[NH_IN], sNwF[D_INNER];

    const int tid  = threadIdx.x;
    const int lane = tid & 31;
    const int wave = tid >> 5;
    const int h    = blockIdx.x / BSZ;
    const int b    = blockIdx.x % BSZ;

    // ---- one-time staging -------------------------------------------------
    for (int idx = tid; idx < NPAD * 16; idx += 256) {
        int j = idx >> 4, k = idx & 15; // j = in_proj output col
        sWinT[j][k] = (j < D_IN_PROJ) ? (_Float16)Win[(h * D_IN_PROJ + j) * 16 + k]
                                      : (_Float16)0.0f;
    }
    for (int idx = tid; idx < 16 * 32; idx += 256) {
        int j = idx >> 5, k = idx & 31;
        sWoT[j][k] = (_Float16)Wout[(h * 16 + j) * 32 + k];
    }
    for (int idx = tid; idx < CONV_DIM * 4; idx += 256) {
        int c = idx >> 2, k = idx & 3;
        sConvW[c][k] = (_Float16)convw[(h * CONV_DIM + c) * 4 + k];
    }
    for (int idx = tid; idx < CONV_DIM; idx += 256) sConvB[idx] = (_Float16)convb[h * CONV_DIM + idx];
    for (int idx = tid; idx < D_INNER; idx += 256)  sNwF[idx]  = nw[h * D_INNER + idx];
    if (tid < NH_IN) {
        sA[tid]   = -__expf(Alog[h * NH_IN + tid]);
        sDtb[tid] = dtb[h * NH_IN + tid];
        sDpv[tid] = Dp[h * NH_IN + tid];
    }
    for (int idx = tid; idx < NH_IN * 16 * D_STATE; idx += 256) ((float*)sHstT)[idx] = 0.0f;
    for (int idx = tid; idx < 3 * CONV_DIM; idx += 256) ((_Float16*)sXtail)[idx] = (_Float16)0.0f;
    __syncthreads();

    // ---- chunk loop -------------------------------------------------------
    for (int ci = 0; ci < NCHUNK; ++ci) {
        const int t0 = ci * CHUNK;

        // stage u; prefetch next chunk (gfx1250 global_prefetch_b8)
        for (int idx = tid; idx < CHUNK * 16; idx += 256) {
            int t = idx >> 4, k = idx & 15;
            sU16[t][k] = (_Float16)x[((size_t)(b * TLEN + t0 + t)) * D_MODEL + h * HEADDIM + k];
        }
        if (ci + 1 < NCHUNK && tid < CHUNK)
            __builtin_prefetch(&x[((size_t)(b * TLEN + t0 + CHUNK + tid)) * D_MODEL + h * HEADDIM], 0, 1);
        __syncthreads();

        // in_proj: [32x16] @ [16x208]  (26 WMMA tiles, K=16 zero-padded)
        for (int job = wave; job < 2 * 13; job += 8) {
            int ti = job / 13, nj = job % 13;
            v16h a  = loadA16_k16(&sU16[ti * 16][0], 16, lane);
            v16h bb = loadBT16_k16(&sWinT[nj * 16][0], 16, lane);
            v8f acc = zero8();
            acc = WMMA_F16(a, bb, acc);
            int n = lane & 15, mb = (lane < 16) ? 0 : 8;
#pragma unroll
            for (int r = 0; r < 8; ++r)
                sZX16[ti * 16 + mb + r][nj * 16 + n] = (_Float16)acc[r];
        }
        __syncthreads();

        // dt softplus + per-step decay factors (parallel)
        for (int idx = tid; idx < NH_IN * CHUNK; idx += 256) {
            int hi = idx & 1, t = idx >> 1;
            float raw = (float)sZX16[t][2 * D_INNER + 2 * D_STATE + hi] + sDtb[hi];
            float d   = softplusf_(raw);
            sdt[hi][t] = d;
            sP[hi][t] = __expf(d * sA[hi]);   // per-step dA, cumprod later
        }
        // depthwise causal conv + silu (reads raw sZX16 + tail)
        for (int idx = tid; idx < CHUNK * CONV_DIM; idx += 256) {
            int t = idx / CONV_DIM, c = idx % CONV_DIM;
            float acc = (float)sConvB[c];
#pragma unroll
            for (int k = 0; k < D_CONV; ++k) {
                int s = t - 3 + k;
                float xv = (s >= 0) ? (float)sZX16[s][D_INNER + c] : (float)sXtail[3 + s][c];
                acc += (float)sConvW[c][k] * xv;
            }
            sConv16[t][c] = (_Float16)siluf_(acc);
        }
        __syncthreads();

        // cumulative product (2 short serial threads) + tail save
        if (tid < NH_IN) {
            float P = 1.0f;
            for (int t = 0; t < CHUNK; ++t) { P *= sP[tid][t]; sP[tid][t] = P; }
            sPl[tid] = P;
        }
        for (int idx = tid; idx < 3 * CONV_DIM; idx += 256) {
            int r = idx / CONV_DIM, c = idx % CONV_DIM;
            sXtail[r][c] = sZX16[CHUNK - 3 + r][D_INNER + c];
        }
        __syncthreads();

        for (int hi = 0; hi < NH_IN; ++hi) {
            // per-hi staging: (dt*x)ᵀ and decayed Bᵀ
            for (int idx = tid; idx < 16 * CHUNK; idx += 256) {
                int p = idx >> 5, s = idx & 31;
                sXdtT[p][s] = (_Float16)(sdt[hi][s] * (float)sConv16[s][hi * 16 + p]);
            }
            for (int idx = tid; idx < D_STATE * CHUNK; idx += 256) {
                int n = idx >> 5, s = idx & 31;
                sBd16[n][s] = (_Float16)((sPl[hi] / sP[hi][s]) * (float)sConv16[s][D_INNER + n]);
            }
            __syncthreads();

            // Stage A (6 wave-jobs): 4x masked G = C·Bᵀ tiles, 2x Y_inter = P_t*(C·h_old)
            for (int job = wave; job < 6; job += 8) {
                if (job < 4) {
                    int ti = job >> 1, tj = job & 1;
                    v8f acc = zero8();
#pragma unroll
                    for (int kk = 0; kk < 2; ++kk) {
                        v16h a  = loadA16(&sConv16[ti * 16][D_INNER + D_STATE + kk * 32], CONV_DIM, lane);
                        v16h bb = loadBT16(&sConv16[tj * 16][D_INNER + kk * 32], CONV_DIM, lane);
                        acc = WMMA_F16(a, bb, acc);
                    }
                    int n = lane & 15, mb = (lane < 16) ? 0 : 8;
#pragma unroll
                    for (int r = 0; r < 8; ++r) {
                        int t = ti * 16 + mb + r, s = tj * 16 + n;
                        float v = (s <= t) ? acc[r] * (sP[hi][t] / sP[hi][s]) : 0.0f;
                        sGm[t][s] = (_Float16)v;
                    }
                } else {
                    int ti = job - 4;
                    v8f acc = zero8();
#pragma unroll
                    for (int kk = 0; kk < 2; ++kk) {
                        v16h a  = loadA16(&sConv16[ti * 16][D_INNER + D_STATE + kk * 32], CONV_DIM, lane);
                        v16h bb = loadBT16f(&sHstT[hi][0][kk * 32], D_STATE, lane);
                        acc = WMMA_F16(a, bb, acc);
                    }
                    int n = lane & 15, mb = (lane < 16) ? 0 : 8;
#pragma unroll
                    for (int r = 0; r < 8; ++r) {
                        int t = ti * 16 + mb + r;
                        sYF[t][hi * 16 + n] = acc[r] * sP[hi][t];
                    }
                }
            }
            __syncthreads();

            // Stage B (6 wave-jobs): 4x state update, 2x Y final (+= Gm·Xdt + D*x)
            for (int job = wave; job < 6; job += 8) {
                if (job < 4) {
                    int mi = job;
                    v16h a  = loadA16(&sBd16[mi * 16][0], 40, lane);
                    v16h bb = loadBT16(&sXdtT[0][0], CHUNK, lane);
                    v8f acc = zero8();
                    acc = WMMA_F16(a, bb, acc);
                    int p = lane & 15, mb = (lane < 16) ? 0 : 8;
#pragma unroll
                    for (int r = 0; r < 8; ++r) {
                        int ns = mi * 16 + mb + r;
                        sHstT[hi][p][ns] = sPl[hi] * sHstT[hi][p][ns] + acc[r];
                    }
                } else {
                    int ti = job - 4;
                    v8f acc = loadC8(&sYF[ti * 16][hi * 16], D_INNER, lane);
                    v16h a  = loadA16(&sGm[ti * 16][0], 40, lane);
                    v16h bb = loadBT16(&sXdtT[0][0], CHUNK, lane);
                    acc = WMMA_F16(a, bb, acc);
                    int n = lane & 15, mb = (lane < 16) ? 0 : 8;
#pragma unroll
                    for (int r = 0; r < 8; ++r) {
                        int t = ti * 16 + mb + r;
                        sYF[t][hi * 16 + n] = acc[r] + sDpv[hi] * (float)sConv16[t][hi * 16 + n];
                    }
                }
            }
            __syncthreads();
        }

        // gate with silu(z), RMSNorm over D_INNER, scale by nw
        if (tid < CHUNK) {
            int t = tid;
            float ss = 0.0f;
            for (int j = 0; j < D_INNER; ++j) {
                float z = (float)sZX16[t][j];
                float y = sYF[t][j] * siluf_(z);
                sYF[t][j] = y;
                ss += y * y;
            }
            float rv = rsqrtf(ss / (float)D_INNER + 1e-5f);
            for (int j = 0; j < D_INNER; ++j)
                sYg16[t][j] = (_Float16)(sYF[t][j] * rv * sNwF[j]);
        }
        __syncthreads();

        // out_proj: [32x32] @ [32x16] -> global (K=32 exact)
        for (int job = wave; job < 2; job += 8) {
            int ti = job;
            v16h a  = loadA16(&sYg16[ti * 16][0], D_INNER, lane);
            v16h bb = loadBT16(&sWoT[0][0], 32, lane);
            v8f acc = zero8();
            acc = WMMA_F16(a, bb, acc);
            int n = lane & 15, mb = (lane < 16) ? 0 : 8;
#pragma unroll
            for (int r = 0; r < 8; ++r) {
                int tg = t0 + ti * 16 + mb + r;
                yout[(((size_t)(h * BSZ + b)) * TLEN + tg) * HEADDIM + n] = acc[r];
            }
        }
        __syncthreads();
    }
}

// ===========================================================================
// Kernel 2: diff = y1 - lam*y2 ; layernorm over (T,HEADDIM) per (h,b);
// apply gn_w/gn_b; write f16 activation matrix [B*T, 256]. grid = NH*BSZ.
// ===========================================================================
__global__ __launch_bounds__(256) void diff_norm_kernel(
    const float* __restrict__ y0, const float* __restrict__ y1,
    const float* __restrict__ lam, const float* __restrict__ gnw,
    const float* __restrict__ gnb, _Float16* __restrict__ omat)
{
    const int tid = threadIdx.x;
    const int h = blockIdx.x / BSZ, b = blockIdx.x % BSZ;
    const size_t base = ((size_t)(h * BSZ + b)) * TLEN * HEADDIM;
    const int N = TLEN * HEADDIM;

    __shared__ float red[256], red2[256], sMean, sRinv;
    float sum = 0.0f, ss = 0.0f;
    for (int i = tid; i < N; i += 256) {
        int j = i & 15;
        float d = y0[base + i] - lam[h * HEADDIM + j] * y1[base + i];
        sum += d; ss += d * d;
    }
    red[tid] = sum; red2[tid] = ss;
    __syncthreads();
    for (int s = 128; s > 0; s >>= 1) {
        if (tid < s) { red[tid] += red[tid + s]; red2[tid] += red2[tid + s]; }
        __syncthreads();
    }
    if (tid == 0) {
        float mean = red[0] / (float)N;
        float var  = red2[0] / (float)N - mean * mean;
        sMean = mean;
        sRinv = rsqrtf(var + 1e-5f);
    }
    __syncthreads();
    for (int i = tid; i < N; i += 256) {
        int t = i >> 4, j = i & 15;
        float d = y0[base + i] - lam[h * HEADDIM + j] * y1[base + i];
        float v = (d - sMean) * sRinv * gnw[h * HEADDIM + j] + gnb[h * HEADDIM + j];
        omat[((size_t)(b * TLEN + t)) * D_MODEL + h * HEADDIM + j] = (_Float16)v;
    }
}

// ===========================================================================
// Kernel 3: out = x + omat @ proj_wᵀ + proj_b.  M=4096, N=256, K=256.
// One 16x16 WMMA tile per wave, K-chained (8 WMMA). grid = 512, block = 256.
// ===========================================================================
__global__ __launch_bounds__(256) void final_proj_kernel(
    const _Float16* __restrict__ omat, const float* __restrict__ proj_w,
    const float* __restrict__ proj_b, const float* __restrict__ x,
    float* __restrict__ out)
{
    __shared__ __align__(16) _Float16 sPw[16][256]; // proj_w rows (BT layout)
    const int tid = threadIdx.x, lane = tid & 31, wave = tid >> 5;
    const int nj = blockIdx.x & 15;
    const int mg = blockIdx.x >> 4; // 0..31

    for (int idx = tid; idx < 16 * 256; idx += 256) {
        int n = idx >> 8, k = idx & 255;
        sPw[n][k] = (_Float16)proj_w[((size_t)(nj * 16 + n)) * 256 + k];
    }
    __syncthreads();

    const int mtile = mg * 8 + wave; // 0..255
    const _Float16* Ap = omat + (size_t)mtile * 16 * 256;
    v8f acc = zero8();
#pragma unroll
    for (int kk = 0; kk < 8; ++kk) {
        v16h a  = loadA16(Ap + kk * 32, 256, lane);
        v16h bb = loadBT16(&sPw[0][kk * 32], 256, lane);
        acc = WMMA_F16(a, bb, acc);
    }
    int n = lane & 15, mb = (lane < 16) ? 0 : 8;
    int col = nj * 16 + n;
#pragma unroll
    for (int r = 0; r < 8; ++r) {
        size_t tok = (size_t)mtile * 16 + mb + r;
        out[tok * 256 + col] = x[tok * 256 + col] + acc[r] + proj_b[col];
    }
}

// ===========================================================================
extern "C" void kernel_launch(void* const* d_in, const int* in_sizes, int n_in,
                              void* d_out, int out_size, void* d_ws, size_t ws_size,
                              hipStream_t stream) {
    const float* x      = (const float*)d_in[0];
    const float* lam    = (const float*)d_in[17];
    const float* gnw    = (const float*)d_in[18];
    const float* gnb    = (const float*)d_in[19];
    const float* proj_w = (const float*)d_in[20];
    const float* proj_b = (const float*)d_in[21];

    const size_t ysz = (size_t)NH * BSZ * TLEN * HEADDIM; // 1,048,576 floats
    float*    y0   = (float*)d_ws;
    float*    y1   = y0 + ysz;
    _Float16* omat = (_Float16*)(y1 + ysz);

    dim3 blk(256);
    mamba_fwd_kernel<<<dim3(NH * BSZ), blk, 0, stream>>>(
        x, (const float*)d_in[1], (const float*)d_in[2], (const float*)d_in[3],
        (const float*)d_in[4], (const float*)d_in[5], (const float*)d_in[6],
        (const float*)d_in[7], (const float*)d_in[8], y0);
    mamba_fwd_kernel<<<dim3(NH * BSZ), blk, 0, stream>>>(
        x, (const float*)d_in[9], (const float*)d_in[10], (const float*)d_in[11],
        (const float*)d_in[12], (const float*)d_in[13], (const float*)d_in[14],
        (const float*)d_in[15], (const float*)d_in[16], y1);
    diff_norm_kernel<<<dim3(NH * BSZ), blk, 0, stream>>>(y0, y1, lam, gnw, gnb, omat);
    final_proj_kernel<<<dim3(512), blk, 0, stream>>>(omat, proj_w, proj_b, x, (float*)d_out);
}
